// OnlineTripletLoss_16423954940178
// MI455X (gfx1250) — matrix operand order repaired and checked by reading, not compiled, not run on hardware
//
#include <hip/hip_runtime.h>
#include <hip/hip_bf16.h>

typedef __attribute__((ext_vector_type(16))) _Float16 v16h;
typedef __attribute__((ext_vector_type(8)))  _Float16 v8h;
typedef __attribute__((ext_vector_type(8)))  float    v8f;

#define N_SAMP 8192
#define DIM    128
#define NCLS   10
#define NTILES (N_SAMP / 16)

// ---------------- workspace layout (bytes) ----------------
#define E16_BYTES   ((size_t)N_SAMP * DIM * 2)
#define SQ_OFF      (E16_BYTES)
#define PERM_OFF    (SQ_OFF + (size_t)N_SAMP * 4)
#define CSTART_OFF  (PERM_OFF + (size_t)N_SAMP * 4)
#define NEG10_OFF   (CSTART_OFF + 64)
#define BSUM_OFF    (NEG10_OFF + 512)

// ---- Kernel 1: squared norms + fp32 -> fp16 conversion ----
__global__ void prep_kernel(const float* __restrict__ emb,
                            _Float16* __restrict__ e16,
                            float* __restrict__ sq) {
  int j = blockIdx.x * blockDim.x + threadIdx.x;
  if (j >= N_SAMP) return;
  const float* row = emb + (size_t)j * DIM;
  _Float16* orow = e16 + (size_t)j * DIM;
  float s = 0.f;
  for (int k = 0; k < DIM; k += 4) {
    float4 v = *(const float4*)(row + k);
    s += v.x * v.x + v.y * v.y + v.z * v.z + v.w * v.w;
    orow[k + 0] = (_Float16)v.x;
    orow[k + 1] = (_Float16)v.y;
    orow[k + 2] = (_Float16)v.z;
    orow[k + 3] = (_Float16)v.w;
  }
  sq[j] = s;
}

// ---- Kernel 2: stable counting sort by label + neg10 table (tiny, label-only) ----
__global__ void sort_kernel(const int* __restrict__ labels,
                            int* __restrict__ perm,
                            int* __restrict__ classStart,
                            int* __restrict__ neg10) {
  if (threadIdx.x != 0) return;
  int cnt[NCLS];
  for (int c = 0; c < NCLS; ++c) cnt[c] = 0;
  for (int i = 0; i < N_SAMP; ++i) cnt[labels[i]]++;
  int start[NCLS + 1];
  start[0] = 0;
  for (int c = 0; c < NCLS; ++c) start[c + 1] = start[c] + cnt[c];
  for (int c = 0; c <= NCLS; ++c) classStart[c] = start[c];
  int pos[NCLS];
  for (int c = 0; c < NCLS; ++c) pos[c] = start[c];
  for (int i = 0; i < N_SAMP; ++i) perm[pos[labels[i]]++] = i;  // stable (class, idx)
  for (int L = 0; L < NCLS; ++L) {
    int k = 0;
    for (int p = 0; p < N_SAMP && k < NCLS; ++p) {
      int idx = perm[p];
      if (labels[idx] != L) neg10[L * NCLS + k++] = idx;
    }
  }
}

// ---- Kernel 3: fused WMMA distance mining + loss (one wave per 16-row tile) ----
__global__ void __launch_bounds__(32)
mine_kernel(const float* __restrict__ emb,
            const int* __restrict__ labels,
            const _Float16* __restrict__ e16,
            const float* __restrict__ sq,
            const int* __restrict__ perm,
            const int* __restrict__ classStart,
            const int* __restrict__ neg10,
            float* __restrict__ blockSums) {
  __shared__ float clsSum[16][NCLS];
  __shared__ int   wpIdx[16];

  const int lane = threadIdx.x;       // 0..31, wave32
  const int hlf  = lane >> 4;
  const int nloc = lane & 15;
  const int i0   = blockIdx.x * 16;

  // A fragments: 16x32 fp16, lane holds row M=lane%16;
  // VGPR0-3 <- K = klo..klo+7, VGPR4-7 <- K = klo+8..klo+15 (klo = 16*half)
  const _Float16* ap = e16 + (size_t)(i0 + nloc) * DIM;
  const int klo = hlf * 16;
  v16h afrag[4];
#pragma unroll
  for (int kb = 0; kb < 4; ++kb) {
    v8h lo = *(const v8h*)(ap + kb * 32 + klo);
    v8h hi = *(const v8h*)(ap + kb * 32 + 8 + klo);
    v16h a;
#pragma unroll
    for (int t = 0; t < 8; ++t) { a[t] = lo[t]; a[8 + t] = hi[t]; }
    afrag[kb] = a;
  }

  float sqi[8]; int rc[8];
#pragma unroll
  for (int r = 0; r < 8; ++r) {
    int row = i0 + r + 8 * hlf;       // C/D layout: elem r -> row r + 8*(lane/16)
    sqi[r] = sq[row];
    rc[r]  = labels[row];
  }

  float segSum[8], bestD[8]; int bestI[8];
#pragma unroll
  for (int r = 0; r < 8; ++r) { segSum[r] = 0.f; bestD[r] = -3.4e38f; bestI[r] = 0; }

  for (int c = 0; c < NCLS; ++c) {
    const int s = __builtin_amdgcn_readfirstlane(classStart[c]);
    const int e = __builtin_amdgcn_readfirstlane(classStart[c + 1]);
    const int jt0 = (s >> 4), jt1 = ((e + 15) >> 4);

    // software-pipeline prologue: index + norm of the first column
    int   colP = jt0 * 16 + nloc;
    int   srcP = perm[colP < N_SAMP ? colP : N_SAMP - 1];
    float sqjP = sq[srcP];

    for (int jt = jt0; jt < jt1; ++jt) {
      const int   col = colP;
      const int   src = srcP;
      const float sqj = sqjP;

      // B fragment: 32x16 fp16, lane holds col N=lane%16;
      // lanes<16 K=0..15, lanes>=16 K=16..31 of each 32-K step.
      // Distinct SSA values -> scheduler can clause all 8 b128 loads and
      // stagger s_wait_loadcnt across the 4 WMMAs.
      const v16h* bp = (const v16h*)(e16 + (size_t)src * DIM);
      const v16h b0 = bp[0 + hlf];
      const v16h b1 = bp[2 + hlf];
      const v16h b2 = bp[4 + hlf];
      const v16h b3 = bp[6 + hlf];

      // prefetch next column's gather index + norm (overlaps the WMMAs)
      colP = col + 16;
      {
        int cl = colP < N_SAMP ? colP : N_SAMP - 1;
        srcP = perm[cl];
        sqjP = sq[srcP];
      }

      v8f acc = {};
      acc = __builtin_amdgcn_wmma_f32_16x16x32_f16(false, afrag[0], false, b0, (short)0, acc, false, false);
      acc = __builtin_amdgcn_wmma_f32_16x16x32_f16(false, afrag[1], false, b1, (short)0, acc, false, false);
      acc = __builtin_amdgcn_wmma_f32_16x16x32_f16(false, afrag[2], false, b2, (short)0, acc, false, false);
      acc = __builtin_amdgcn_wmma_f32_16x16x32_f16(false, afrag[3], false, b3, (short)0, acc, false, false);

      // Branchless epilogue: selects only, EXEC stays all-1s
      const bool valid = (col >= s) & (col < e);
#pragma unroll
      for (int r = 0; r < 8; ++r) {
        float Dv = sqi[r] + sqj - 2.0f * acc[r];   // squared distance
        segSum[r] += valid ? Dv : 0.0f;            // per-class column sum
        bool take = valid & (rc[r] == c) & (Dv > bestD[r]);
        bestD[r] = take ? Dv  : bestD[r];
        bestI[r] = take ? src : bestI[r];
      }
    }
    // reduce segment sums across the 16-lane half (rows fixed per half)
#pragma unroll
    for (int r = 0; r < 8; ++r) {
      float v = segSum[r];
      v += __shfl_xor(v, 1, 32);
      v += __shfl_xor(v, 2, 32);
      v += __shfl_xor(v, 4, 32);
      v += __shfl_xor(v, 8, 32);
      if (nloc == 0) clsSum[r + 8 * hlf][c] = v;
      segSum[r] = 0.f;
    }
  }

  // hardest-positive argmax reduce across the 16-lane half (branchless combine)
#pragma unroll
  for (int r = 0; r < 8; ++r) {
    float d = bestD[r]; int ix = bestI[r];
#pragma unroll
    for (int m = 1; m < 16; m <<= 1) {
      float od = __shfl_xor(d, m, 32);
      int   oi = __shfl_xor(ix, m, 32);
      bool better = (od > d) | ((od == d) & (oi < ix));
      d  = better ? od : d;
      ix = better ? oi : ix;
    }
    if (nloc == 0) wpIdx[r + 8 * hlf] = ix;
  }
  __syncthreads();

  // Final per-row: argmin over neg_dist, buggy neg10 lookup, exact fp32 loss
  float partial = 0.f;
  if (lane < 16) {
    const int m = lane, i = i0 + m;
    float cs[NCLS], tot = 0.f;
#pragma unroll
    for (int c = 0; c < NCLS; ++c) { cs[c] = clsSum[m][c]; tot += cs[c]; }
    int jstar = 0; float best = tot - cs[0];
#pragma unroll
    for (int c = 1; c < NCLS; ++c) {
      float v = tot - cs[c];
      bool lt = v < best;
      best  = lt ? v : best;
      jstar = lt ? c : jstar;
    }
    const int lab = labels[i];
    const int wn  = neg10[lab * NCLS + jstar];
    const int wp  = wpIdx[m];
    const float* ei = emb + (size_t)i  * DIM;
    const float* ep = emb + (size_t)wp * DIM;
    const float* en = emb + (size_t)wn * DIM;
    float dp = 0.f, dn = 0.f;
    for (int k = 0; k < DIM; k += 4) {
      float4 a = *(const float4*)(ei + k);
      float4 b = *(const float4*)(ep + k);
      float4 g = *(const float4*)(en + k);
      float dx;
      dx = a.x - b.x; dp += dx * dx;  dx = a.y - b.y; dp += dx * dx;
      dx = a.z - b.z; dp += dx * dx;  dx = a.w - b.w; dp += dx * dx;
      dx = a.x - g.x; dn += dx * dx;  dx = a.y - g.y; dn += dx * dx;
      dx = a.z - g.z; dn += dx * dx;  dx = a.w - g.w; dn += dx * dx;
    }
    float l = dp - dn + 1.0f;          // MARGIN
    partial = l > 0.f ? l : 0.f;
  }
  partial += __shfl_xor(partial, 1, 32);
  partial += __shfl_xor(partial, 2, 32);
  partial += __shfl_xor(partial, 4, 32);
  partial += __shfl_xor(partial, 8, 32);
  partial += __shfl_xor(partial, 16, 32);
  if (lane == 0) blockSums[blockIdx.x] = partial;
}

// ---- Kernel 4: deterministic final reduction ----
__global__ void reduce_kernel(const float* __restrict__ blockSums,
                              float* __restrict__ out) {
  if (threadIdx.x != 0) return;
  float s = 0.f;
  for (int b = 0; b < NTILES; ++b) s += blockSums[b];
  out[0] = s;
}

extern "C" void kernel_launch(void* const* d_in, const int* in_sizes, int n_in,
                              void* d_out, int out_size, void* d_ws, size_t ws_size,
                              hipStream_t stream) {
  const float* emb    = (const float*)d_in[0];
  const int*   labels = (const int*)d_in[1];
  char* ws = (char*)d_ws;
  _Float16* e16      = (_Float16*)ws;
  float*    sq       = (float*)(ws + SQ_OFF);
  int*      perm     = (int*)(ws + PERM_OFF);
  int*      cstart   = (int*)(ws + CSTART_OFF);
  int*      neg10    = (int*)(ws + NEG10_OFF);
  float*    bsums    = (float*)(ws + BSUM_OFF);
  float*    out      = (float*)d_out;

  prep_kernel<<<N_SAMP / 256, 256, 0, stream>>>(emb, e16, sq);
  sort_kernel<<<1, 32, 0, stream>>>(labels, perm, cstart, neg10);
  mine_kernel<<<NTILES, 32, 0, stream>>>(emb, labels, e16, sq, perm, cstart, neg10, bsums);
  reduce_kernel<<<1, 32, 0, stream>>>(bsums, out);
}
